// Gemma3ForMultimodalLMwithTPA_30760555774687
// MI455X (gfx1250) — matrix operand order repaired and checked by reading, not compiled, not run on hardware
//
#include <hip/hip_runtime.h>
#include <cmath>

// ---------------------------------------------------------------------------
// Gemma3 TPA attention layer for MI455X (gfx1250, wave32)
// bf16 WMMA everywhere + Tensor Data Mover (TDM) double-buffered LDS staging.
// ---------------------------------------------------------------------------

typedef __attribute__((ext_vector_type(16))) __bf16 v16bf;
typedef __attribute__((ext_vector_type(8)))  float  v8f;
typedef __attribute__((ext_vector_type(4)))  unsigned int v4u;
typedef __attribute__((ext_vector_type(8)))  int    v8i;
typedef __attribute__((ext_vector_type(4)))  int    v4i;

constexpr int Bc = 2, Sc = 2048, Hc = 2048, NHc = 8, HDc = 256;
constexpr int QRc = 6, KRc = 2, VRc = 2;
constexpr int Mrows = Bc * Sc;                 // 4096
constexpr float SCALINGc = 0.0625f;            // 256^-0.5
constexpr float SOFTCAPc = 50.0f;
constexpr float EPSc = 1e-6f;
constexpr float NEGINFc = -2.3819763e+38f;

// ---------------------------------------------------------------------------
// TDM 2D tile load: global (row-major, bf16) -> LDS, with optional LDS row
// padding. Descriptor per CDNA5 ISA 8.3/8.4 (D# group0/group1).
//   pad_interval code c: pad after 2^(c+1) DWORDs; pad_amount code a: a+1 DWORDs
// OOB rows/cols (beyond tensor_d0/tensor_d1) read as zero -> free edge guard.
// Must be called wave-uniformly (one wave per workgroup issues it).
// ---------------------------------------------------------------------------
__device__ __forceinline__ void tdm_load_2d(unsigned lds_addr, const void* gaddr,
                                            unsigned tile_d0, unsigned tile_d1,
                                            unsigned tensor_d0, unsigned tensor_d1,
                                            unsigned stride0,
                                            unsigned pad_int_code, unsigned pad_amt_code,
                                            bool pad_en) {
    unsigned long long ga = (unsigned long long)gaddr;
    v4u g0;
    g0.x = 1u;                                           // count=1 (user descriptor)
    g0.y = lds_addr;                                     // LDS byte address
    g0.z = (unsigned)(ga & 0xffffffffu);                 // global_addr[31:0]
    g0.w = (unsigned)((ga >> 32) & 0x01ffffffu) | (2u << 30);  // addr[56:32] | type=2
    unsigned d0 = (1u << 16);                            // data_size=1 -> 2 bytes
    if (pad_en) d0 |= (1u << 20) | (pad_int_code << 22) | (pad_amt_code << 25);
    v8i g1;
    g1[0] = (int)d0;                                     // wg_mask=0, flags
    g1[1] = (int)((tensor_d0 & 0xffffu) << 16);          // tensor_dim0[15:0] @ bit48
    g1[2] = (int)((tensor_d0 >> 16) | ((tensor_d1 & 0xffffu) << 16));
    g1[3] = (int)((tensor_d1 >> 16) | (tile_d0 << 16));  // tile_dim0 @ bit112
    g1[4] = (int)(tile_d1 & 0xffffu);                    // tile_dim1; tile_dim2=0
    g1[5] = (int)stride0;                                // tensor_dim0_stride[31:0]
    g1[6] = 0;                                           // stride0[47:32], stride1 lo
    g1[7] = 0;
    v4i z4 = {0, 0, 0, 0};
#if __clang_major__ >= 23
    v8i z8v = {0, 0, 0, 0, 0, 0, 0, 0};
    __builtin_amdgcn_tensor_load_to_lds(g0, g1, z4, z4, z8v, 0);
#else
    __builtin_amdgcn_tensor_load_to_lds(g0, g1, z4, z4, 0);
#endif
}

__device__ __forceinline__ unsigned lds_off(const void* p) {
    // low 32 bits of the flat address of a __shared__ object == LDS byte offset
    return (unsigned)(unsigned long long)(uintptr_t)p;
}

// ---------------------------------------------------------------------------
// WMMA fragment loaders (wave32; contiguous k = 16*(lane/16) + s, same order in A and B)
// ---------------------------------------------------------------------------
__device__ __forceinline__ v16bf frag_a(const __bf16* __restrict__ base, int stride, int lane) {
    int m = lane & 15, hi = lane >> 4;
    const __bf16* row = base + m * stride;
    v16bf f;
#pragma unroll
    for (int p = 0; p < 8; ++p) {
        int k = hi * 16 + 2 * p;
        f[2 * p]     = row[k];
        f[2 * p + 1] = row[k + 1];
    }
    return f;
}

__device__ __forceinline__ v16bf frag_b_nmajor(const __bf16* __restrict__ base, int stride, int lane) {
    int n = lane & 15, hi = lane >> 4;
    const __bf16* col = base + n * stride;
    v16bf f;
#pragma unroll
    for (int p = 0; p < 8; ++p) {
        int k = hi * 16 + 2 * p;
        f[2 * p]     = col[k];
        f[2 * p + 1] = col[k + 1];
    }
    return f;
}

__device__ __forceinline__ v16bf frag_b_kmajor(const __bf16* __restrict__ base, int stride, int lane) {
    int n = lane & 15, hi = lane >> 4;
    v16bf f;
#pragma unroll
    for (int p = 0; p < 8; ++p) {
        int k = hi * 16 + 2 * p;
        f[2 * p]     = base[(size_t)k * stride + n];
        f[2 * p + 1] = base[(size_t)(k + 1) * stride + n];
    }
    return f;
}

// ---------------------------------------------------------------------------
// fp32 -> bf16 conversion (grid-stride)
// ---------------------------------------------------------------------------
__global__ void cvt_bf16(const float* __restrict__ s, __bf16* __restrict__ d, long long n) {
    long long i  = (long long)blockIdx.x * blockDim.x + threadIdx.x;
    long long st = (long long)gridDim.x * blockDim.x;
    for (; i < n; i += st) d[i] = (__bf16)s[i];
}

// ---------------------------------------------------------------------------
// bf16 WMMA GEMM: C[M,N] (fp32) = A[M,K] (bf16 row-major) * Bw[N,K]^T
// 128x64 tile per 256-thread block; wave = 32x32 (4 WMMA / K-step of 32).
// TDM double-buffered LDS staging: DMA of tile t+1 overlaps WMMA of tile t.
// ---------------------------------------------------------------------------
__global__ void __launch_bounds__(256)
gemm_bf16(const __bf16* __restrict__ A, const __bf16* __restrict__ Bw,
          float* __restrict__ C, int Mn, int Nn, int Kn) {
    constexpr int TS = 40;                          // 32 + 4-DWORD TDM pad
    __shared__ __bf16 As[2][128 * TS];
    __shared__ __bf16 Bs[2][64 * TS];

    int tid = threadIdx.x, lane = tid & 31, wid = tid >> 5;
    int m0 = blockIdx.y * 128, n0 = blockIdx.x * 64;
    int mo = (wid & 3) * 32, no = (wid >> 2) * 32;

    v8f c00 = {}, c01 = {}, c10 = {}, c11 = {};

    if (wid == 0) {                                 // prologue: DMA tile 0
        tdm_load_2d(lds_off(&As[0][0]), &A[(size_t)m0 * Kn], 32, 128,
                    (unsigned)Kn, (unsigned)(Mn - m0), (unsigned)Kn, 3, 3, true);
        tdm_load_2d(lds_off(&Bs[0][0]), &Bw[(size_t)n0 * Kn], 32, 64,
                    (unsigned)Kn, (unsigned)(Nn - n0), (unsigned)Kn, 3, 3, true);
        __builtin_amdgcn_s_wait_tensorcnt(0);
    }
    __syncthreads();

    int nsteps = Kn >> 5;
    for (int s = 0; s < nsteps; ++s) {
        int cur = s & 1, nxt = cur ^ 1;
        if (wid == 0 && s + 1 < nsteps) {           // DMA next tile while computing
            int k0 = (s + 1) * 32;
            tdm_load_2d(lds_off(&As[nxt][0]), &A[(size_t)m0 * Kn + k0], 32, 128,
                        (unsigned)(Kn - k0), (unsigned)(Mn - m0), (unsigned)Kn, 3, 3, true);
            tdm_load_2d(lds_off(&Bs[nxt][0]), &Bw[(size_t)n0 * Kn + k0], 32, 64,
                        (unsigned)(Kn - k0), (unsigned)(Nn - n0), (unsigned)Kn, 3, 3, true);
        }
        const __bf16* as = As[cur];
        const __bf16* bs = Bs[cur];
        v16bf a0 = frag_a(&as[(mo + 0) * TS], TS, lane);
        v16bf a1 = frag_a(&as[(mo + 16) * TS], TS, lane);
        v16bf b0 = frag_b_nmajor(&bs[(no + 0) * TS], TS, lane);
        v16bf b1 = frag_b_nmajor(&bs[(no + 16) * TS], TS, lane);
        c00 = __builtin_amdgcn_wmma_f32_16x16x32_bf16(false, a0, false, b0, (short)0, c00, false, false);
        c01 = __builtin_amdgcn_wmma_f32_16x16x32_bf16(false, a0, false, b1, (short)0, c01, false, false);
        c10 = __builtin_amdgcn_wmma_f32_16x16x32_bf16(false, a1, false, b0, (short)0, c10, false, false);
        c11 = __builtin_amdgcn_wmma_f32_16x16x32_bf16(false, a1, false, b1, (short)0, c11, false, false);
        if (wid == 0) __builtin_amdgcn_s_wait_tensorcnt(0);
        __syncthreads();
    }

    int nn = lane & 15, mh = (lane >> 4) * 8;
#pragma unroll
    for (int r = 0; r < 8; ++r) {
        int row0 = m0 + mo + mh + r, row1 = row0 + 16;
        int col0 = n0 + no + nn, col1 = col0 + 16;
        if (col0 < Nn) {
            C[(size_t)row0 * Nn + col0] = c00[r];
            C[(size_t)row1 * Nn + col0] = c10[r];
        }
        if (col1 < Nn) {
            C[(size_t)row0 * Nn + col1] = c01[r];
            C[(size_t)row1 * Nn + col1] = c11[r];
        }
    }
}

// ---------------------------------------------------------------------------
// RoPE in-place on Bm [Mrows, R, HD] fp32, half-split pairing (d, d+128)
// ---------------------------------------------------------------------------
__global__ void rope_kernel(float* __restrict__ Bm, const float* __restrict__ fc,
                            const float* __restrict__ fs, int R) {
    int i = blockIdx.x * blockDim.x + threadIdx.x;
    int total = Mrows * R * (HDc / 2);
    if (i >= total) return;
    int d = i & 127;
    int t = i >> 7;
    int r = t % R;
    int m = t / R;
    int s = m % Sc;
    float c = fc[s * 128 + d], sn = fs[s * 128 + d];
    size_t base = (size_t)m * R * HDc + (size_t)r * HDc;
    float x1 = Bm[base + d], x2 = Bm[base + d + 128];
    Bm[base + d]       = x1 * c - x2 * sn;
    Bm[base + d + 128] = x1 * sn + x2 * c;
}

// ---------------------------------------------------------------------------
// Rank contraction + RMSNorm; writes q/k/v bf16 in [B, NH, S, HD].
// ---------------------------------------------------------------------------
__global__ void __launch_bounds__(256)
combine_kernel(const float* __restrict__ Aq, const float* __restrict__ Ak,
               const float* __restrict__ Av, const float* __restrict__ Bq,
               const float* __restrict__ Bk, const float* __restrict__ Bv,
               const float* __restrict__ qnw, const float* __restrict__ knw,
               __bf16* __restrict__ qo, __bf16* __restrict__ ko, __bf16* __restrict__ vo) {
    int m = blockIdx.x;
    int lane = threadIdx.x & 31, n = threadIdx.x >> 5;
    int b = m / Sc, s = m - b * Sc;
    size_t obase = ((size_t)(b * NHc + n) * Sc + s) * (size_t)HDc;

    {   // q: QR ranks, RMSNorm(qnorm), * SCALING
        float a[QRc];
#pragma unroll
        for (int r = 0; r < QRc; ++r) a[r] = Aq[(size_t)m * (NHc * QRc) + n * QRc + r];
        float val[8], ss = 0.f;
#pragma unroll
        for (int j = 0; j < 8; ++j) {
            int d = lane + 32 * j;
            float acc = 0.f;
#pragma unroll
            for (int r = 0; r < QRc; ++r)
                acc += a[r] * Bq[(size_t)m * (QRc * HDc) + r * HDc + d];
            acc *= (1.0f / QRc);
            val[j] = acc; ss += acc * acc;
        }
#pragma unroll
        for (int msk = 16; msk >= 1; msk >>= 1) ss += __shfl_xor(ss, msk, 32);
        float rms = rsqrtf(ss * (1.0f / HDc) + EPSc);
#pragma unroll
        for (int j = 0; j < 8; ++j) {
            int d = lane + 32 * j;
            qo[obase + d] = (__bf16)(val[j] * rms * (1.0f + qnw[d]) * SCALINGc);
        }
    }
    {   // k: KR ranks, RMSNorm(knorm)
        float a[KRc];
#pragma unroll
        for (int r = 0; r < KRc; ++r) a[r] = Ak[(size_t)m * (NHc * KRc) + n * KRc + r];
        float val[8], ss = 0.f;
#pragma unroll
        for (int j = 0; j < 8; ++j) {
            int d = lane + 32 * j;
            float acc = 0.f;
#pragma unroll
            for (int r = 0; r < KRc; ++r)
                acc += a[r] * Bk[(size_t)m * (KRc * HDc) + r * HDc + d];
            acc *= (1.0f / KRc);
            val[j] = acc; ss += acc * acc;
        }
#pragma unroll
        for (int msk = 16; msk >= 1; msk >>= 1) ss += __shfl_xor(ss, msk, 32);
        float rms = rsqrtf(ss * (1.0f / HDc) + EPSc);
#pragma unroll
        for (int j = 0; j < 8; ++j) {
            int d = lane + 32 * j;
            ko[obase + d] = (__bf16)(val[j] * rms * (1.0f + knw[d]));
        }
    }
    {   // v: VR ranks
        float a[VRc];
#pragma unroll
        for (int r = 0; r < VRc; ++r) a[r] = Av[(size_t)m * (NHc * VRc) + n * VRc + r];
#pragma unroll
        for (int j = 0; j < 8; ++j) {
            int d = lane + 32 * j;
            float acc = 0.f;
#pragma unroll
            for (int r = 0; r < VRc; ++r)
                acc += a[r] * Bv[(size_t)m * (VRc * HDc) + r * HDc + d];
            vo[obase + d] = (__bf16)(acc * (1.0f / VRc));
        }
    }
}

// ---------------------------------------------------------------------------
// Flash-style causal attention with tanh softcap.
// grid (S/128, B*NH); 8 waves, wave w owns q rows [q0+16w, +16).
// K/V chunks (32 keys x 256 d) streamed by TDM with double buffering.
// Dynamic LDS: Ks[2] | Vs[2] | Ps  (TDM pads rows 256 -> 264 bf16).
// ---------------------------------------------------------------------------
constexpr int KSTa = 264;
constexpr int PSTa = 40;
constexpr size_t ATTN_LDS_BYTES = (size_t)(4 * 32 * KSTa + 8 * 16 * PSTa) * 2;

__global__ void __launch_bounds__(256)
attn_kernel(const __bf16* __restrict__ Q, const __bf16* __restrict__ K,
            const __bf16* __restrict__ V, __bf16* __restrict__ O) {
    extern __shared__ __bf16 smem[];
    __bf16* KsBuf = smem;                     // [2][32*KSTa]
    __bf16* VsBuf = smem + 2 * 32 * KSTa;     // [2][32*KSTa]
    __bf16* Ps    = smem + 4 * 32 * KSTa;     // [8][16*PSTa]

    int tid = threadIdx.x, lane = tid & 31, wid = tid >> 5;
    int bh = blockIdx.y;
    int b = bh >> 3, h = bh & 7;
    int q0 = blockIdx.x * 128;
    size_t hbase = (size_t)bh * Sc * HDc;
    int qrow0 = q0 + wid * 16;
    int nn = lane & 15, mh = (lane >> 4) * 8;

    // resident Q fragments: 8 chunks of 16x32 bf16
    v16bf qf[8];
    {
        int mloc = lane & 15, hi = lane >> 4;
        const __bf16* qrow = Q + hbase + (size_t)(qrow0 + mloc) * HDc;
#pragma unroll
        for (int kk = 0; kk < 8; ++kk) {
#pragma unroll
            for (int p = 0; p < 8; ++p) {
                int k = kk * 32 + hi * 16 + 2 * p;
                qf[kk][2 * p]     = qrow[k];
                qf[kk][2 * p + 1] = qrow[k + 1];
            }
        }
    }

    v8f o[16];
    v8f z8 = {};
#pragma unroll
    for (int t = 0; t < 16; ++t) o[t] = z8;
    float rmax[8], rsum[8];
#pragma unroll
    for (int r = 0; r < 8; ++r) { rmax[r] = NEGINFc; rsum[r] = 0.f; }

    __bf16* Pw = &Ps[wid * 16 * PSTa];
    int nchunks = (q0 + 128) >> 5;            // causal: keys < q0+128

    if (wid == 0) {                           // prologue: DMA chunk 0
        tdm_load_2d(lds_off(KsBuf), K + hbase, 256, 32, 256, (unsigned)Sc, 256, 6, 3, true);
        tdm_load_2d(lds_off(VsBuf), V + hbase, 256, 32, 256, (unsigned)Sc, 256, 6, 3, true);
        __builtin_amdgcn_s_wait_tensorcnt(0);
    }
    __syncthreads();

    for (int ci = 0; ci < nchunks; ++ci) {
        int kb = ci * 32;
        int cur = ci & 1, nxt = cur ^ 1;
        if (wid == 0 && ci + 1 < nchunks) {   // DMA next chunk while computing
            size_t goff = hbase + (size_t)(kb + 32) * HDc;
            tdm_load_2d(lds_off(KsBuf + nxt * 32 * KSTa), K + goff, 256, 32,
                        256, (unsigned)(Sc - kb - 32), 256, 6, 3, true);
            tdm_load_2d(lds_off(VsBuf + nxt * 32 * KSTa), V + goff, 256, 32,
                        256, (unsigned)(Sc - kb - 32), 256, 6, 3, true);
        }
        const __bf16* Ks = KsBuf + cur * 32 * KSTa;
        const __bf16* Vs = VsBuf + cur * 32 * KSTa;

        // scores: 16x32 per wave
        v8f sc0 = z8, sc1 = z8;
#pragma unroll
        for (int kk = 0; kk < 8; ++kk) {
            v16bf kb0 = frag_b_nmajor(&Ks[0 * KSTa + kk * 32], KSTa, lane);
            v16bf kb1 = frag_b_nmajor(&Ks[16 * KSTa + kk * 32], KSTa, lane);
            sc0 = __builtin_amdgcn_wmma_f32_16x16x32_bf16(false, qf[kk], false, kb0, (short)0, sc0, false, false);
            sc1 = __builtin_amdgcn_wmma_f32_16x16x32_bf16(false, qf[kk], false, kb1, (short)0, sc1, false, false);
        }

        // softcap + causal mask + online softmax
        float pn[2][8], curmax[8];
#pragma unroll
        for (int r = 0; r < 8; ++r) curmax[r] = NEGINFc;
#pragma unroll
        for (int nt = 0; nt < 2; ++nt) {
#pragma unroll
            for (int r = 0; r < 8; ++r) {
                float sv = (nt == 0) ? sc0[r] : sc1[r];
                sv = SOFTCAPc * tanhf(sv * (1.0f / SOFTCAPc));
                int qr = qrow0 + mh + r;
                int kc = kb + nt * 16 + nn;
                if (kc > qr) sv = NEGINFc;
                pn[nt][r] = sv;
                curmax[r] = fmaxf(curmax[r], sv);
            }
        }
#pragma unroll
        for (int msk = 8; msk >= 1; msk >>= 1)
#pragma unroll
            for (int r = 0; r < 8; ++r)
                curmax[r] = fmaxf(curmax[r], __shfl_xor(curmax[r], msk, 32));

        float alpha[8], psum[8];
#pragma unroll
        for (int r = 0; r < 8; ++r) {
            float nm = fmaxf(rmax[r], curmax[r]);
            alpha[r] = __expf(rmax[r] - nm);
            rmax[r] = nm;
            psum[r] = 0.f;
        }
#pragma unroll
        for (int nt = 0; nt < 2; ++nt) {
#pragma unroll
            for (int r = 0; r < 8; ++r) {
                float p = __expf(pn[nt][r] - rmax[r]);
                Pw[(mh + r) * PSTa + nt * 16 + nn] = (__bf16)p;
                psum[r] += p;
            }
        }
#pragma unroll
        for (int msk = 8; msk >= 1; msk >>= 1)
#pragma unroll
            for (int r = 0; r < 8; ++r)
                psum[r] += __shfl_xor(psum[r], msk, 32);
#pragma unroll
        for (int r = 0; r < 8; ++r) rsum[r] = rsum[r] * alpha[r] + psum[r];

#pragma unroll
        for (int t = 0; t < 16; ++t)
#pragma unroll
            for (int r = 0; r < 8; ++r) o[t][r] *= alpha[r];

        v16bf pa = frag_a(Pw, PSTa, lane);
#pragma unroll
        for (int t = 0; t < 16; ++t) {
            v16bf vb = frag_b_kmajor(&Vs[t * 16], KSTa, lane);
            o[t] = __builtin_amdgcn_wmma_f32_16x16x32_bf16(false, pa, false, vb, (short)0, o[t], false, false);
        }
        if (wid == 0) __builtin_amdgcn_s_wait_tensorcnt(0);
        __syncthreads();
    }

#pragma unroll
    for (int r = 0; r < 8; ++r) {
        float inv = 1.0f / rsum[r];
        size_t row = (size_t)(b * Sc + qrow0 + mh + r);
#pragma unroll
        for (int t = 0; t < 16; ++t) {
            O[row * (size_t)(NHc * HDc) + h * HDc + t * 16 + nn] = (__bf16)(o[t][r] * inv);
        }
    }
}

// ---------------------------------------------------------------------------
// Host launcher
// ---------------------------------------------------------------------------
extern "C" void kernel_launch(void* const* d_in, const int* in_sizes, int n_in,
                              void* d_out, int out_size, void* d_ws, size_t ws_size,
                              hipStream_t stream) {
    (void)in_sizes; (void)n_in; (void)out_size; (void)ws_size;
    const float* x   = (const float*)d_in[0];
    const float* fc  = (const float*)d_in[1];
    const float* fs  = (const float*)d_in[2];
    // d_in[3] kv_write_indices, d_in[4] mask: causal mask computed inline
    const float* wAq = (const float*)d_in[5];
    const float* wAk = (const float*)d_in[6];
    const float* wAv = (const float*)d_in[7];
    const float* wBq = (const float*)d_in[8];
    const float* wBk = (const float*)d_in[9];
    const float* wBv = (const float*)d_in[10];
    const float* wo  = (const float*)d_in[11];
    const float* qnw = (const float*)d_in[12];
    const float* knw = (const float*)d_in[13];
    float* out = (float*)d_out;

    char* ws = (char*)d_ws;
    size_t off = 0;
    auto alloc = [&](size_t bytes) { size_t o = off; off = (off + bytes + 255) & ~(size_t)255; return o; };

    __bf16* xb    = (__bf16*)(ws + alloc((size_t)Mrows * Hc * 2));
    __bf16* wAqb  = (__bf16*)(ws + alloc((size_t)48 * Hc * 2));
    __bf16* wAkb  = (__bf16*)(ws + alloc((size_t)16 * Hc * 2));
    __bf16* wAvb  = (__bf16*)(ws + alloc((size_t)16 * Hc * 2));
    __bf16* wBqb  = (__bf16*)(ws + alloc((size_t)1536 * Hc * 2));
    __bf16* wBkb  = (__bf16*)(ws + alloc((size_t)512 * Hc * 2));
    __bf16* wBvb  = (__bf16*)(ws + alloc((size_t)512 * Hc * 2));
    __bf16* wob   = (__bf16*)(ws + alloc((size_t)2048 * Hc * 2));
    float*  Aqp   = (float*)(ws + alloc((size_t)Mrows * 48 * 4));
    float*  Akp   = (float*)(ws + alloc((size_t)Mrows * 16 * 4));
    float*  Avp   = (float*)(ws + alloc((size_t)Mrows * 16 * 4));
    float*  Bqp   = (float*)(ws + alloc((size_t)Mrows * 1536 * 4));
    float*  Bkp   = (float*)(ws + alloc((size_t)Mrows * 512 * 4));
    float*  Bvp   = (float*)(ws + alloc((size_t)Mrows * 512 * 4));
    __bf16* qws   = (__bf16*)(ws + alloc((size_t)Bc * NHc * Sc * HDc * 2));
    __bf16* kws   = (__bf16*)(ws + alloc((size_t)Bc * NHc * Sc * HDc * 2));
    __bf16* vws   = (__bf16*)(ws + alloc((size_t)Bc * NHc * Sc * HDc * 2));
    __bf16* attnb = (__bf16*)(ws + alloc((size_t)Mrows * (NHc * HDc) * 2));

    // 1) bf16 conversions
    cvt_bf16<<<512, 256, 0, stream>>>(x, xb, (long long)Mrows * Hc);
    cvt_bf16<<<64, 256, 0, stream>>>(wAq, wAqb, 48LL * Hc);
    cvt_bf16<<<64, 256, 0, stream>>>(wAk, wAkb, 16LL * Hc);
    cvt_bf16<<<64, 256, 0, stream>>>(wAv, wAvb, 16LL * Hc);
    cvt_bf16<<<256, 256, 0, stream>>>(wBq, wBqb, 1536LL * Hc);
    cvt_bf16<<<128, 256, 0, stream>>>(wBk, wBkb, 512LL * Hc);
    cvt_bf16<<<128, 256, 0, stream>>>(wBv, wBvb, 512LL * Hc);
    cvt_bf16<<<512, 256, 0, stream>>>(wo, wob, 2048LL * Hc);

    // 2) projection GEMMs (TDM-staged bf16 WMMA)
    gemm_bf16<<<dim3(1, Mrows / 128), 256, 0, stream>>>(xb, wAqb, Aqp, Mrows, 48, Hc);
    gemm_bf16<<<dim3(1, Mrows / 128), 256, 0, stream>>>(xb, wAkb, Akp, Mrows, 16, Hc);
    gemm_bf16<<<dim3(1, Mrows / 128), 256, 0, stream>>>(xb, wAvb, Avp, Mrows, 16, Hc);
    gemm_bf16<<<dim3(1536 / 64, Mrows / 128), 256, 0, stream>>>(xb, wBqb, Bqp, Mrows, 1536, Hc);
    gemm_bf16<<<dim3(512 / 64, Mrows / 128), 256, 0, stream>>>(xb, wBkb, Bkp, Mrows, 512, Hc);
    gemm_bf16<<<dim3(512 / 64, Mrows / 128), 256, 0, stream>>>(xb, wBvb, Bvp, Mrows, 512, Hc);

    // 3) RoPE on B_q / B_k
    {
        int tq = Mrows * QRc * (HDc / 2);
        rope_kernel<<<(tq + 255) / 256, 256, 0, stream>>>(Bqp, fc, fs, QRc);
        int tk = Mrows * KRc * (HDc / 2);
        rope_kernel<<<(tk + 255) / 256, 256, 0, stream>>>(Bkp, fc, fs, KRc);
    }

    // 4) rank contraction + RMSNorm -> q/k/v bf16 [B,NH,S,HD]
    combine_kernel<<<Mrows, 256, 0, stream>>>(Aqp, Akp, Avp, Bqp, Bkp, Bvp,
                                              qnw, knw, qws, kws, vws);

    // 5) flash attention with softcap (TDM K/V streaming + WMMA)
    hipFuncSetAttribute((const void*)attn_kernel,
                        hipFuncAttributeMaxDynamicSharedMemorySize,
                        (int)ATTN_LDS_BYTES);
    attn_kernel<<<dim3(Sc / 128, Bc * NHc), 256, ATTN_LDS_BYTES, stream>>>(qws, kws, vws, attnb);

    // 6) output projection -> fp32 d_out
    gemm_bf16<<<dim3(Hc / 64, Mrows / 128), 256, 0, stream>>>(attnb, wob, out, Mrows, Hc, Hc);
}
